// FixedPointSelfAttentionStep_2027224564463
// MI455X (gfx1250) — matrix operand order repaired and checked by reading, not compiled, not run on hardware
//
#include <hip/hip_runtime.h>
#include <hip/hip_bf16.h>

typedef _Float16 v16h __attribute__((ext_vector_type(16)));
typedef _Float16 v8h  __attribute__((ext_vector_type(8)));
typedef float    v8f  __attribute__((ext_vector_type(8)));
typedef int      v4i  __attribute__((ext_vector_type(4)));

#define WMMA_F32_F16(a, b, c) \
  __builtin_amdgcn_wmma_f32_16x16x32_f16(false, (a), false, (b), (short)0, (c), false, false)

#if defined(__has_builtin)
#if __has_builtin(__builtin_amdgcn_global_load_async_to_lds_b128)
#define HAVE_ASYNC_LDS 1
#endif
#if __has_builtin(__builtin_amdgcn_s_wait_asynccnt)
#define HAVE_WAIT_ASYNC 1
#endif
#endif

static __device__ __forceinline__ void async_copy16(const float* g, float* l) {
#ifdef HAVE_ASYNC_LDS
  float* gn = const_cast<float*>(g);
  __builtin_amdgcn_global_load_async_to_lds_b128(
      (__attribute__((address_space(1))) v4i*)gn,
      (__attribute__((address_space(3))) v4i*)l, 0, 0);
#else
  *(float4*)l = *(const float4*)g;   // synchronous fallback
#endif
}

#ifdef HAVE_WAIT_ASYNC
#define WAIT_ASYNC(n) __builtin_amdgcn_s_wait_asynccnt(n)
#elif defined(HAVE_ASYNC_LDS)
#define WAIT_ASYNC(n) asm volatile("s_wait_asynccnt " #n ::: "memory")
#else
#define WAIT_ASYNC(n) do {} while (0)
#endif

static __device__ __forceinline__ v16h cat8(v8h lo, v8h hi) {
  return __builtin_shufflevector(lo, hi, 0,1,2,3,4,5,6,7,8,9,10,11,12,13,14,15);
}
static __device__ __forceinline__ v8h ld8(const _Float16* p) { return *(const v8h*)p; }

// Build a 16-half A/B fragment from two runs of 8 consecutive f32 in LDS.
static __device__ __forceinline__ v16h cvt16(const float* p0, const float* p1) {
  float4 a = *(const float4*)p0, b = *(const float4*)(p0 + 4);
  float4 c = *(const float4*)p1, d = *(const float4*)(p1 + 4);
  v16h r;
  r[0]=(_Float16)a.x;  r[1]=(_Float16)a.y;  r[2]=(_Float16)a.z;  r[3]=(_Float16)a.w;
  r[4]=(_Float16)b.x;  r[5]=(_Float16)b.y;  r[6]=(_Float16)b.z;  r[7]=(_Float16)b.w;
  r[8]=(_Float16)c.x;  r[9]=(_Float16)c.y;  r[10]=(_Float16)c.z; r[11]=(_Float16)c.w;
  r[12]=(_Float16)d.x; r[13]=(_Float16)d.y; r[14]=(_Float16)d.z; r[15]=(_Float16)d.w;
  return r;
}

// ---------------------------------------------------------------------------
// Kernel 1: qkv = z_k @ Wqkv^T + b; keep only Q (cols 0..1023) and K (1024..2047),
// store as f16 in [B, H, N, D]. Block tile 64x64, 8 waves.
// Double-buffered GLOBAL_LOAD_ASYNC_TO_LDS pipeline (ASYNCcnt tracked).
// ---------------------------------------------------------------------------
__global__ __launch_bounds__(256) void qk_gemm_kernel(
    const float* __restrict__ zk, const float* __restrict__ W,
    const float* __restrict__ bias,
    _Float16* __restrict__ Qh, _Float16* __restrict__ Kh)
{
  __shared__ float Af[2][64 * 36];   // 64 rows x 32 k, stride 36 f32 (16B aligned, bank-spread)
  __shared__ float Bf[2][64 * 36];
  const int tid  = threadIdx.x;
  const int lane = tid & 31;
  const int wave = tid >> 5;
  const int lrow = lane & 15;
  const int half = lane >> 4;
  const int m0 = blockIdx.x * 64;   // row block of flattened (B*N)=8192
  const int j0 = blockIdx.y * 64;   // col block of 2048 (q|k)
  const int r0 = (wave & 3) * 16;   // wave's row sub-tile
  const int c0 = (wave >> 2) * 32;  // wave's col sub-tile (2x16)
  const int flr = tid >> 2;         // cooperative-fill row 0..63
  const int flc = (tid & 3) * 8;    // cooperative-fill col base (floats)

  const float* ga0 = zk + (size_t)(m0 + flr) * 1024 + flc;
  const float* gb0 = W  + (size_t)(j0 + flr) * 1024 + flc;

  v8f acc0 = {}; v8f acc1 = {};

  // prologue: async-fill buffer 0 with k-tile 0 (4 b128 ops per thread)
  async_copy16(ga0,     &Af[0][flr * 36 + flc]);
  async_copy16(ga0 + 4, &Af[0][flr * 36 + flc + 4]);
  async_copy16(gb0,     &Bf[0][flr * 36 + flc]);
  async_copy16(gb0 + 4, &Bf[0][flr * 36 + flc + 4]);

  for (int it = 0; it < 32; ++it) {
    const int buf = it & 1;
    if (it + 1 < 32) {
      const int k1 = (it + 1) * 32;
      async_copy16(ga0 + k1,     &Af[buf ^ 1][flr * 36 + flc]);
      async_copy16(ga0 + k1 + 4, &Af[buf ^ 1][flr * 36 + flc + 4]);
      async_copy16(gb0 + k1,     &Bf[buf ^ 1][flr * 36 + flc]);
      async_copy16(gb0 + k1 + 4, &Bf[buf ^ 1][flr * 36 + flc + 4]);
      WAIT_ASYNC(4);   // current buffer's 4 fills retired (in-order), next 4 in flight
    } else {
      WAIT_ASYNC(0);
    }
    __syncthreads();   // every wave waited -> whole tile visible

    // A fragment (16x32): lane group holds contiguous K half {0..15} / {16..31}
    const float* arow = &Af[buf][(r0 + lrow) * 36 + (half ? 16 : 0)];
    v16h af = cvt16(arow, arow + 8);
    // B fragments: lane group holds contiguous K half {0..15} / {16..31}
    const float* br0 = &Bf[buf][(c0 + lrow) * 36 + (half ? 16 : 0)];
    const float* br1 = &Bf[buf][(c0 + 16 + lrow) * 36 + (half ? 16 : 0)];
    acc0 = WMMA_F32_F16(af, cvt16(br0, br0 + 8), acc0);
    acc1 = WMMA_F32_F16(af, cvt16(br1, br1 + 8), acc1);
    __syncthreads();   // all waves done reading buf before it is refilled
  }

  // C/D layout: VGPR r, lane<16 -> (M=r, N=lane); lane>=16 -> (M=8+r, N=lane-16)
  #pragma unroll
  for (int ch = 0; ch < 2; ++ch) {
    const int j = j0 + c0 + ch * 16 + lrow;
    const float bj = bias[j];
    _Float16* dst = (j < 1024) ? Qh : Kh;
    const int jj = j & 1023;
    const int h = jj >> 6, d = jj & 63;
    const v8f acc = ch ? acc1 : acc0;
    #pragma unroll
    for (int r = 0; r < 8; ++r) {
      const int m = m0 + r0 + r + (half ? 8 : 0);
      const int bidx = m >> 10, n = m & 1023;
      dst[((size_t)((bidx * 16 + h) * 1024 + n)) * 64 + d] = (_Float16)(acc[r] + bj);
    }
  }
}

// ---------------------------------------------------------------------------
// Kernel 2: Vt[b,h,d,n] = (f16) x[b,n,h*64+d]  (LDS transpose for coalescing)
// ---------------------------------------------------------------------------
__global__ __launch_bounds__(256) void vt_kernel(
    const float* __restrict__ x, _Float16* __restrict__ Vt)
{
  __shared__ _Float16 T[64 * 72];
  const int tid = threadIdx.x;
  const int b  = blockIdx.x >> 8;
  const int h  = (blockIdx.x >> 4) & 15;
  const int n0 = (blockIdx.x & 15) << 6;
  const int i  = tid >> 2;          // local n, 0..63
  const int jb = (tid & 3) * 16;    // local d base
  const float4* xp = (const float4*)(x + ((size_t)(b * 1024 + n0 + i)) * 1024 + h * 64 + jb);
  float4 x0 = xp[0], x1 = xp[1], x2 = xp[2], x3 = xp[3];
  v8h t0, t1;
  t0[0]=(_Float16)x0.x; t0[1]=(_Float16)x0.y; t0[2]=(_Float16)x0.z; t0[3]=(_Float16)x0.w;
  t0[4]=(_Float16)x1.x; t0[5]=(_Float16)x1.y; t0[6]=(_Float16)x1.z; t0[7]=(_Float16)x1.w;
  t1[0]=(_Float16)x2.x; t1[1]=(_Float16)x2.y; t1[2]=(_Float16)x2.z; t1[3]=(_Float16)x2.w;
  t1[4]=(_Float16)x3.x; t1[5]=(_Float16)x3.y; t1[6]=(_Float16)x3.z; t1[7]=(_Float16)x3.w;
  *(v8h*)&T[i * 72 + jb]     = t0;
  *(v8h*)&T[i * 72 + jb + 8] = t1;
  __syncthreads();
  const int d  = tid >> 2;          // local d, 0..63
  const int nb = (tid & 3) * 16;    // local n base
  v8h o0, o1;
  #pragma unroll
  for (int ii = 0; ii < 8; ++ii) {
    o0[ii] = T[(nb + ii) * 72 + d];
    o1[ii] = T[(nb + 8 + ii) * 72 + d];
  }
  _Float16* dst = Vt + ((size_t)((b * 16 + h) * 64 + d)) * 1024 + n0 + nb;
  *(v8h*)dst       = o0;
  *(v8h*)(dst + 8) = o1;
}

// ---------------------------------------------------------------------------
// Kernel 3: causal flash attention per (b,h): S = Q K^T * scale, online softmax,
// O = P V, tanh, store f32. 128 rows/block, 16 rows/wave, 32-key tiles.
// ---------------------------------------------------------------------------
__global__ __launch_bounds__(256) void attn_kernel(
    const _Float16* __restrict__ Qh, const _Float16* __restrict__ Kh,
    const _Float16* __restrict__ Vt, const float* __restrict__ temp,
    float* __restrict__ out)
{
  __shared__ _Float16 Pl[8 * 16 * 72];  // per-wave 16x32 P scratch (stride 72)
  const int lane = threadIdx.x & 31;
  const int wave = threadIdx.x >> 5;
  const int lrow = lane & 15;
  const int half = lane >> 4;
  const int bh = blockIdx.x >> 3;
  const int q0 = (blockIdx.x & 7) * 128;
  const int b = bh >> 4, h = bh & 15;
  const int qr = q0 + wave * 16;

  const _Float16* Qb = Qh + (size_t)bh * 1024 * 64;
  const _Float16* Kb = Kh + (size_t)bh * 1024 * 64;
  const _Float16* Vb = Vt + (size_t)bh * 64 * 1024;
  const float scl = 0.125f / temp[h];  // D^-0.5 / temperature

  // Q fragments (A-layout), kept in registers for whole key loop
  const _Float16* qp = Qb + (size_t)(qr + lrow) * 64 + (half ? 16 : 0);
  v16h qf0 = cat8(ld8(qp),      ld8(qp + 8));
  v16h qf1 = cat8(ld8(qp + 32), ld8(qp + 40));

  float rmax[8], rsum[8];
  v8f o0 = {}, o1 = {}, o2 = {}, o3 = {};
  #pragma unroll
  for (int r = 0; r < 8; ++r) { rmax[r] = -__builtin_inff(); rsum[r] = 0.f; }

  _Float16* pw = &Pl[wave * 16 * 72];
  const int kst = half ? 16 : 0;

  for (int j0 = 0; j0 < qr + 16; j0 += 32) {
    const _Float16* kp  = Kb + (size_t)(j0 + lrow) * 64 + kst;
    const _Float16* kp1 = kp + 16 * 64;
    const _Float16* vp  = Vb + (size_t)lrow * 1024 + j0 + kst;

    // speculative prefetch of the next key tile (global_prefetch_b8)
    __builtin_prefetch(kp + 32 * 64, 0, 1);
    __builtin_prefetch(vp + 32, 0, 1);

    // ---- S = Q K^T for 16 rows x 32 keys (two f32 C fragments) ----
    v8f s0 = {}, s1 = {};
    s0 = WMMA_F32_F16(qf0, cat8(ld8(kp),       ld8(kp + 8)),   s0);
    s0 = WMMA_F32_F16(qf1, cat8(ld8(kp + 32),  ld8(kp + 40)),  s0);
    s1 = WMMA_F32_F16(qf0, cat8(ld8(kp1),      ld8(kp1 + 8)),  s1);
    s1 = WMMA_F32_F16(qf1, cat8(ld8(kp1 + 32), ld8(kp1 + 40)), s1);

    // ---- scale, causal mask, online softmax (rows span 16-lane halves) ----
    const bool needmask = (j0 + 31 > qr);
    #pragma unroll
    for (int r = 0; r < 8; ++r) {
      const int m = qr + r + (half ? 8 : 0);
      float v0 = s0[r] * scl;
      float v1 = s1[r] * scl;
      if (needmask) {
        if (j0 + lrow > m)      v0 = -__builtin_inff();
        if (j0 + 16 + lrow > m) v1 = -__builtin_inff();
      }
      float tm = fmaxf(v0, v1);
      #pragma unroll
      for (int mk = 1; mk < 16; mk <<= 1) tm = fmaxf(tm, __shfl_xor(tm, mk, 32));
      const float nm   = fmaxf(rmax[r], tm);
      const float corr = __expf(rmax[r] - nm);
      rmax[r] = nm;
      const float p0 = __expf(v0 - nm);
      const float p1 = __expf(v1 - nm);
      float ts = p0 + p1;
      #pragma unroll
      for (int mk = 1; mk < 16; mk <<= 1) ts += __shfl_xor(ts, mk, 32);
      rsum[r] = rsum[r] * corr + ts;
      o0[r] *= corr; o1[r] *= corr; o2[r] *= corr; o3[r] *= corr;
      // spill P (C-layout) to wave-private LDS as f16
      const int prow = r + (half ? 8 : 0);
      pw[prow * 72 + lrow]      = (_Float16)p0;
      pw[prow * 72 + 16 + lrow] = (_Float16)p1;
    }

    // ---- reload P in A-layout (16x32), then O += P @ V (4 d-groups) ----
    const _Float16* pr = pw + lrow * 72 + (half ? 16 : 0);
    v16h pf = cat8(*(const v8h*)pr, *(const v8h*)(pr + 8));

    o0 = WMMA_F32_F16(pf, cat8(ld8(vp),             ld8(vp + 8)),             o0);
    o1 = WMMA_F32_F16(pf, cat8(ld8(vp + 16 * 1024), ld8(vp + 16 * 1024 + 8)), o1);
    o2 = WMMA_F32_F16(pf, cat8(ld8(vp + 32 * 1024), ld8(vp + 32 * 1024 + 8)), o2);
    o3 = WMMA_F32_F16(pf, cat8(ld8(vp + 48 * 1024), ld8(vp + 48 * 1024 + 8)), o3);
  }

  // ---- normalize, tanh, store ----
  #pragma unroll
  for (int r = 0; r < 8; ++r) {
    const float inv = 1.0f / rsum[r];
    const int m = qr + r + (half ? 8 : 0);
    float* op = out + ((size_t)(b * 1024 + m)) * 1024 + h * 64 + lrow;
    op[0]  = tanhf(o0[r] * inv);
    op[16] = tanhf(o1[r] * inv);
    op[32] = tanhf(o2[r] * inv);
    op[48] = tanhf(o3[r] * inv);
  }
}

// ---------------------------------------------------------------------------
extern "C" void kernel_launch(void* const* d_in, const int* in_sizes, int n_in,
                              void* d_out, int out_size, void* d_ws, size_t ws_size,
                              hipStream_t stream) {
  (void)in_sizes; (void)n_in; (void)out_size; (void)ws_size;
  const float* zk   = (const float*)d_in[0];
  const float* x    = (const float*)d_in[1];
  const float* W    = (const float*)d_in[2];
  const float* bias = (const float*)d_in[3];
  const float* temp = (const float*)d_in[4];
  float* out = (float*)d_out;

  const size_t QK_ELEMS = (size_t)8 * 16 * 1024 * 64;  // 8M halves = 16 MB each
  _Float16* Qh = (_Float16*)d_ws;
  _Float16* Kh = Qh + QK_ELEMS;
  _Float16* Vt = Kh + QK_ELEMS;

  qk_gemm_kernel<<<dim3(128, 32), 256, 0, stream>>>(zk, W, bias, Qh, Kh);
  vt_kernel<<<2048, 256, 0, stream>>>(x, Vt);
  attn_kernel<<<1024, 256, 0, stream>>>(Qh, Kh, Vt, temp, out);
}
